// Attention_22187801051652
// MI455X (gfx1250) — matrix operand order repaired and checked by reading, not compiled, not run on hardware
//
#include <hip/hip_runtime.h>
#include <hip/hip_bf16.h>

// ---------------------------------------------------------------------------
// Types matching CDNA5 WMMA operand layouts
// ---------------------------------------------------------------------------
typedef __attribute__((ext_vector_type(16))) __bf16 v16bf;
typedef __attribute__((ext_vector_type(8)))  float  v8f;

union BF16x16 { v16bf v; uint4 u[2]; __bf16 e[16]; };
union BF16x8  { uint4 u; __bf16 e[8]; };

#define BM 128
#define BN 128
#define BK 64
#define LDK_P 72  // padded LDS row stride (elems): 144B row, 16B-aligned, bank-conflict-free
#define STAGE_ELEMS (BM * LDK_P)   // 9216 elems = 18432 B per stage buffer

// ---------------------------------------------------------------------------
// fp32 -> bf16 elementwise cast, 8 elems/thread (uint4 packed stores)
// ---------------------------------------------------------------------------
__global__ __launch_bounds__(256) void cvt8_f32_bf16(const float* __restrict__ in,
                                                     __bf16* __restrict__ out, int n8) {
  int i = blockIdx.x * blockDim.x + threadIdx.x;
  const int stride = gridDim.x * blockDim.x;
  for (; i < n8; i += stride) {
    float4 a = ((const float4*)in)[2 * i];
    float4 b = ((const float4*)in)[2 * i + 1];
    BF16x8 o;
    o.e[0] = (__bf16)a.x; o.e[1] = (__bf16)a.y; o.e[2] = (__bf16)a.z; o.e[3] = (__bf16)a.w;
    o.e[4] = (__bf16)b.x; o.e[5] = (__bf16)b.y; o.e[6] = (__bf16)b.z; o.e[7] = (__bf16)b.w;
    ((uint4*)out)[i] = o.u;
  }
}

// ---------------------------------------------------------------------------
// fp32 (K x N) -> bf16 transposed (N x K), LDS-tiled so both sides coalesce
// block = 256 threads (32 x 8), tile 32 x 32
// ---------------------------------------------------------------------------
__global__ __launch_bounds__(256) void transpose_cvt(const float* __restrict__ in,
                                                     __bf16* __restrict__ out,
                                                     int K, int N) {
  __shared__ __bf16 tile[32][33];
  const int kb = blockIdx.y * 32, nb = blockIdx.x * 32;
  const int tx = threadIdx.x & 31, ty = threadIdx.x >> 5;
#pragma unroll
  for (int i = ty; i < 32; i += 8)
    tile[i][tx] = (__bf16)in[(size_t)(kb + i) * N + nb + tx];
  __syncthreads();
#pragma unroll
  for (int i = ty; i < 32; i += 8)
    out[(size_t)(nb + i) * K + kb + tx] = tile[tx][i];
}

// ---------------------------------------------------------------------------
// All-bf16 WMMA GEMM with async LDS staging:
//   C[M,N] = A[M,K] @ BT[N,K]^T   (BT is pre-transposed B)
// 256 threads = 8 waves, wave tile 32(M) x 64(N), BK=64 (2 WMMA K-steps/stage)
// Global->LDS staging uses global_load_async_to_lds_b128 (ASYNCcnt),
// one barrier per K-block, LDS-staged coalesced epilogue.
// Stage layout inside smem: [A0 | A1 | B0 | B1], each STAGE_ELEMS bf16.
// ---------------------------------------------------------------------------
template <bool OUT_F32>
__global__ __launch_bounds__(256) void gemm_wmma_async(const __bf16* __restrict__ A,
                                                       const __bf16* __restrict__ BT,
                                                       void* __restrict__ Cptr,
                                                       const float* __restrict__ bias,
                                                       int M, int N, int K) {
  // 4 stage buffers = 73728 B; epilogue staging (<= 64 KB) reuses this space.
  __shared__ __align__(16) __bf16 smem[4 * STAGE_ELEMS];

  const int tid  = threadIdx.x;
  const int lane = tid & 31;
  const int wid  = tid >> 5;
  const int l15  = lane & 15;
  const int hi   = lane >> 4;

  const int rowBase = blockIdx.y * BM;
  const int colBase = blockIdx.x * BN;
  const int wm = (wid & 3) * 32;
  const int wn = (wid >> 2) * 64;

  // async-copy mapping: thread owns 64 B of one row in each tile (4 x b128)
  const int cRow = tid >> 1;          // 0..127
  const int cK0  = (tid & 1) * 32;    // element offset in K

  v8f acc[2][4];
#pragma unroll
  for (int i = 0; i < 2; i++)
#pragma unroll
    for (int j = 0; j < 4; j++) acc[i][j] = {};

  const int NKB = K / BK;

  auto issueTile = [&](int kb, int buf) {
    const __bf16* Ag = A + (size_t)(rowBase + cRow) * K + kb * BK + cK0;
    const __bf16* Bg = BT + (size_t)(colBase + cRow) * K + kb * BK + cK0;
    unsigned la = (unsigned)(size_t)(smem + buf * STAGE_ELEMS + cRow * LDK_P + cK0);
    unsigned lb = (unsigned)(size_t)(smem + (2 + buf) * STAGE_ELEMS + cRow * LDK_P + cK0);
#pragma unroll
    for (int j = 0; j < 4; ++j) {
      asm volatile("global_load_async_to_lds_b128 %0, %1, off"
                   :: "v"(la + j * 16),
                      "v"((unsigned long long)(size_t)(Ag + j * 8))
                   : "memory");
      asm volatile("global_load_async_to_lds_b128 %0, %1, off"
                   :: "v"(lb + j * 16),
                      "v"((unsigned long long)(size_t)(Bg + j * 8))
                   : "memory");
    }
  };

  issueTile(0, 0);
  asm volatile("s_wait_asynccnt 0x0" ::: "memory");
  __syncthreads();

  for (int kb = 0; kb < NKB; ++kb) {
    const int buf = kb & 1;
    if (kb + 1 < NKB) issueTile(kb + 1, buf ^ 1);  // fills other buffer under compute

    const __bf16* stA = smem + buf * STAGE_ELEMS;
    const __bf16* stB = smem + (2 + buf) * STAGE_ELEMS;

#pragma unroll
    for (int ks = 0; ks < BK; ks += 32) {
      BF16x16 afr[2], bfr[4];
#pragma unroll
      for (int i = 0; i < 2; i++) {
        // A frag (16x32): lane l15 = row; chunks at k = ks+8*hi and ks+16+8*hi
        const __bf16* p = stA + (wm + i * 16 + l15) * LDK_P + ks;
        afr[i].u[0] = *(const uint4*)(p + 8 * hi);
        afr[i].u[1] = *(const uint4*)(p + 16 + 8 * hi);
      }
#pragma unroll
      for (int j = 0; j < 4; j++) {
        // B frag (32x16): lane l15 = col(n); chunks at k = ks+8*hi and ks+16+8*hi
        const __bf16* p = stB + (wn + j * 16 + l15) * LDK_P + ks + 8 * hi;
        bfr[j].u[0] = *(const uint4*)(p);
        bfr[j].u[1] = *(const uint4*)(p + 16);
      }
#pragma unroll
      for (int i = 0; i < 2; i++)
#pragma unroll
        for (int j = 0; j < 4; j++)
          acc[i][j] = __builtin_amdgcn_wmma_f32_16x16x32_bf16(
              false, afr[i].v, false, bfr[j].v, (short)0, acc[i][j], false, false);
    }

    if (kb + 1 < NKB) asm volatile("s_wait_asynccnt 0x0" ::: "memory");
    __syncthreads();
  }

  // ---------------- epilogue: stage tile in LDS, then coalesced b128 stores --
  if (OUT_F32) {
    float* epi = (float*)smem;  // 128*128*4 = 65536 B <= 73728 B
#pragma unroll
    for (int i = 0; i < 2; i++) {
      const int r0 = wm + i * 16 + 8 * hi;
#pragma unroll
      for (int j = 0; j < 4; j++) {
        const int n0 = wn + j * 16 + l15;
        const float bv = bias ? bias[colBase + n0] : 0.f;
#pragma unroll
        for (int r = 0; r < 8; r++) epi[(r0 + r) * BN + n0] = acc[i][j][r] + bv;
      }
    }
    __syncthreads();
    const int row = tid >> 1, c0 = (tid & 1) * 64;
    float* Cg = (float*)Cptr + (size_t)(rowBase + row) * N + colBase + c0;
    const float* src = epi + row * BN + c0;
#pragma unroll
    for (int v = 0; v < 64; v += 4) *(float4*)(Cg + v) = *(const float4*)(src + v);
  } else {
    __bf16* epi = (__bf16*)smem;  // 128*128*2 = 32768 B
#pragma unroll
    for (int i = 0; i < 2; i++) {
      const int r0 = wm + i * 16 + 8 * hi;
#pragma unroll
      for (int j = 0; j < 4; j++) {
        const int n0 = wn + j * 16 + l15;
#pragma unroll
        for (int r = 0; r < 8; r++) epi[(r0 + r) * BN + n0] = (__bf16)acc[i][j][r];
      }
    }
    __syncthreads();
    const int row = tid >> 1, c0 = (tid & 1) * 64;
    __bf16* Cg = (__bf16*)Cptr + (size_t)(rowBase + row) * N + colBase + c0;
    const __bf16* src = epi + row * BN + c0;
#pragma unroll
    for (int v = 0; v < 64; v += 8) *(uint4*)(Cg + v) = *(const uint4*)(src + v);
  }
}

// ---------------------------------------------------------------------------
// Per-token head-attention: one wave per token (16 heads x 64 dim).
//   S = scale * q k^T (2 WMMA), row softmax (shfl over 16-lane halves),
//   O = P v (4 WMMA, K zero-padded), O written in (B,H,N,D) scrambled layout
// ---------------------------------------------------------------------------
__global__ __launch_bounds__(256) void attn_heads(const __bf16* __restrict__ qkv,
                                                  __bf16* __restrict__ obuf) {
  __shared__ float  ldsP[8][16 * 16];
  __shared__ __bf16 ldsV[8][16 * 64];

  const int tid = threadIdx.x, lane = tid & 31, wid = tid >> 5;
  const int l15 = lane & 15, hi = lane >> 4;
  const int tok = blockIdx.x * 8 + wid;  // 0..32767
  const int b = tok >> 12, n = tok & 4095;
  const __bf16* row = qkv + (size_t)tok * 3072;

  {  // stage V (16x64 bf16 = 2KB) into LDS, 64B per lane
    const __bf16* vs = row + 2048 + lane * 32;
    uint4* dst = (uint4*)&ldsV[wid][lane * 32];
    dst[0] = *(const uint4*)(vs);
    dst[1] = *(const uint4*)(vs + 8);
    dst[2] = *(const uint4*)(vs + 16);
    dst[3] = *(const uint4*)(vs + 24);
  }

  v8f s = {};
#pragma unroll
  for (int ks = 0; ks < 64; ks += 32) {
    BF16x16 a, bk;
    const __bf16* qp = row + l15 * 64 + ks;  // q row = head l15
    a.u[0] = *(const uint4*)(qp + 8 * hi);
    a.u[1] = *(const uint4*)(qp + 16 + 8 * hi);
    const __bf16* kp = row + 1024 + l15 * 64 + ks + 8 * hi;  // k row = head l15
    bk.u[0] = *(const uint4*)(kp);
    bk.u[1] = *(const uint4*)(kp + 16);
    s = __builtin_amdgcn_wmma_f32_16x16x32_bf16(false, a.v, false, bk.v, (short)0, s,
                                                false, false);
  }

  // softmax over columns (axis g lives across the 16-lane half)
  float p[8], mx[8], sm[8];
#pragma unroll
  for (int r = 0; r < 8; r++) { p[r] = s[r] * 0.125f; mx[r] = p[r]; }  // 1/sqrt(64)
#pragma unroll
  for (int m = 1; m < 16; m <<= 1)
#pragma unroll
    for (int r = 0; r < 8; r++) mx[r] = fmaxf(mx[r], __shfl_xor(mx[r], m, 32));
#pragma unroll
  for (int r = 0; r < 8; r++) { p[r] = __expf(p[r] - mx[r]); sm[r] = p[r]; }
#pragma unroll
  for (int m = 1; m < 16; m <<= 1)
#pragma unroll
    for (int r = 0; r < 8; r++) sm[r] += __shfl_xor(sm[r], m, 32);
#pragma unroll
  for (int r = 0; r < 8; r++) p[r] /= sm[r];

#pragma unroll
  for (int r = 0; r < 8; r++) ldsP[wid][(r + 8 * hi) * 16 + l15] = p[r];
  __syncthreads();

  BF16x16 ap;
  {
    const float* pp = &ldsP[wid][l15 * 16 + 8 * hi];
#pragma unroll
    for (int i = 0; i < 8; i++) ap.e[i] = (__bf16)pp[i];
#pragma unroll
    for (int i = 0; i < 8; i++) ap.e[8 + i] = (__bf16)0.f;
  }
#pragma unroll
  for (int c = 0; c < 4; c++) {
    const int d = c * 16 + l15;
    BF16x16 bv;
    {
#pragma unroll
      for (int i = 0; i < 8; i++) bv.e[i] = ldsV[wid][(8 * hi + i) * 64 + d];  // K = head g
#pragma unroll
      for (int i = 0; i < 8; i++) bv.e[8 + i] = (__bf16)0.f;  // K 16..31 pad
    }
    v8f o = {};
    o = __builtin_amdgcn_wmma_f32_16x16x32_bf16(false, ap.v, false, bv.v, (short)0, o,
                                                false, false);
#pragma unroll
    for (int r = 0; r < 8; r++) {
      const int m = r + 8 * hi;
      obuf[(((size_t)b * 16 + m) * 4096 + n) * 64 + d] = (__bf16)o[r];
    }
  }
}

// ---------------------------------------------------------------------------
// launch
// ---------------------------------------------------------------------------
extern "C" void kernel_launch(void* const* d_in, const int* in_sizes, int n_in,
                              void* d_out, int out_size, void* d_ws, size_t ws_size,
                              hipStream_t stream) {
  const float* x      = (const float*)d_in[0];  // (8,4096,1024)
  const float* w_qkv  = (const float*)d_in[1];  // (1024,3072)
  const float* w_proj = (const float*)d_in[2];  // (1024,1024)
  const float* b_proj = (const float*)d_in[3];  // (1024,)
  float* out = (float*)d_out;

  const int M = 8 * 4096;  // 32768 tokens
  char* ws = (char*)d_ws;
  __bf16* xb     = (__bf16*)(ws);                    // 32768*1024*2 = 67,108,864
  __bf16* wqkvT  = (__bf16*)(ws + 67108864ULL);      // 3072*1024*2  =  6,291,456
  __bf16* wprojT = (__bf16*)(ws + 73400320ULL);      // 1024*1024*2  =  2,097,152
  __bf16* qkvb   = (__bf16*)(ws + 75497472ULL);      // 32768*3072*2 = 201,326,592
  __bf16* obuf   = (__bf16*)(ws + 276824064ULL);     // 32768*1024*2 = 67,108,864

  // 1) casts + weight transposes
  cvt8_f32_bf16<<<4096, 256, 0, stream>>>(x, xb, (M * 1024) / 8);
  transpose_cvt<<<dim3(3072 / 32, 1024 / 32), 256, 0, stream>>>(w_qkv, wqkvT, 1024, 3072);
  transpose_cvt<<<dim3(1024 / 32, 1024 / 32), 256, 0, stream>>>(w_proj, wprojT, 1024, 1024);

  // 2) QKV = x @ w_qkv  (bf16 out)
  gemm_wmma_async<false><<<dim3(3072 / BN, M / BM), 256, 0, stream>>>(
      xb, wqkvT, (void*)qkvb, nullptr, M, 3072, 1024);

  // 3) per-token 16x16 head attention -> o in (B,H,N,D) layout
  attn_heads<<<M / 8, 256, 0, stream>>>(qkvb, obuf);

  // 4) out = o @ w_proj + b_proj  (fp32 out)
  gemm_wmma_async<true><<<dim3(1024 / BN, M / BM), 256, 0, stream>>>(
      obuf, wprojT, (void*)out, b_proj, M, 1024, 1024);
}
